// SpectralConv2d_48275432407349
// MI455X (gfx1250) — matrix operand order repaired and checked by reading, not compiled, not run on hardware
//
#include <hip/hip_runtime.h>
#include <hip/hip_bf16.h>

typedef __attribute__((ext_vector_type(16))) _Float16 v16h;
typedef __attribute__((ext_vector_type(8)))  _Float16 v8h;
typedef __attribute__((ext_vector_type(8)))  float    v8f;

#define LANE (threadIdx.x & 31)
#define WAVE (threadIdx.x >> 5)

// ---------- WMMA wrapper (codegen-confirmed builtin, 8 args) ----------
static __device__ __forceinline__ v8f wmma32(v16h a, v16h b, v8f c) {
  return __builtin_amdgcn_wmma_f32_16x16x32_f16(false, a, false, b, (short)0, c,
                                                false, false);
}

// A fragment (16x32 f16) from row-major f16 src; per-lane: row = lane&15,
// K = 0..15 (lanes 0-15) or 16..31 (lanes 16-31). Two 16B loads.
static __device__ __forceinline__ v16h ldA16(const _Float16* A, int row0, int k0, int ld) {
  int lane = LANE;
  const _Float16* p = A + (row0 + (lane & 15)) * ld + k0 + ((lane & 16) ? 16 : 0);
  v8h lo = *(const v8h*)p;
  v8h hi = *(const v8h*)(p + 8);
  v16h a;
#pragma unroll
  for (int j = 0; j < 8; ++j) { a[j] = lo[j]; a[8 + j] = hi[j]; }
  return a;
}

// Same A fragment but source is row-major f32 (converted on load).
static __device__ __forceinline__ v16h ldA32(const float* A, int row0, int k0, int ld) {
  int lane = LANE;
  const float* p = A + (row0 + (lane & 15)) * ld + k0 + ((lane & 16) ? 16 : 0);
  v16h a;
#pragma unroll
  for (int j = 0; j < 8; ++j) { a[j] = (_Float16)p[j]; a[8 + j] = (_Float16)p[8 + j]; }
  return a;
}

// B fragment (32x16 f16) from B-transposed storage Bt[n][k]; per-lane:
// col n = lane&15, K = k0 + (lane<16 ? 0..15 : 16..31). Two 16B loads.
static __device__ __forceinline__ v16h ldBT(const _Float16* Bt, int n0, int k0, int ld) {
  int lane = LANE;
  const _Float16* p = Bt + (n0 + (lane & 15)) * ld + k0 + ((lane & 16) ? 16 : 0);
  v8h lo = *(const v8h*)p;
  v8h hi = *(const v8h*)(p + 8);
  v16h b;
#pragma unroll
  for (int j = 0; j < 8; ++j) { b[j] = lo[j]; b[8 + j] = hi[j]; }
  return b;
}

// ---------- init: twiddle tables (exact angles via mod-256) ----------
__global__ __launch_bounds__(256) void init_twiddles(_Float16* T1t, _Float16* A2,
                                                     _Float16* A4, _Float16* T5) {
  int idx = blockIdx.x * 256 + threadIdx.x;  // 16384
  const float C = 0.02454369260617026f;      // 2*pi/256
  {  // T1t[n][k] (64x256): stage-1 B^T: cols 0..19 cos, 20..39 -sin
    int n = idx >> 8, k = idx & 255;
    float v = 0.f;
    if (n < 20) v = cosf(C * (float)((n * k) & 255));
    else if (n < 40) v = -sinf(C * (float)(((n - 20) * k) & 255));
    T1t[idx] = (_Float16)v;
  }
  {  // A2[r][h] (64x256): rows 0..19 cos(kx*h), rows 32..51 sin(kx*h)
    int r = idx >> 8, h = idx & 255;
    float v = 0.f;
    if (r < 20) v = cosf(C * (float)((r * h) & 255));
    else if (r >= 32 && r < 52) v = sinf(C * (float)(((r - 32) * h) & 255));
    A2[idx] = (_Float16)v;
  }
  {  // A4[h][c] (256x64): cols 0..19 cos(kx*h), 32..51 sin(kx*h)
    int h = idx >> 6, c = idx & 63;
    float v = 0.f;
    if (c < 20) v = cosf(C * (float)((c * h) & 255));
    else if (c >= 32 && c < 52) v = sinf(C * (float)(((c - 32) * h) & 255));
    A4[idx] = (_Float16)v;
  }
  {  // T5[w][k] (256x64): stage-5 B^T: k=0 ->1, 1..19 ->2cos, 21..39 ->-2sin
    int w = idx >> 6, k = idx & 63;
    float v = 0.f;
    if (k == 0) v = 1.f;
    else if (k < 20) v = 2.f * cosf(C * (float)((k * w) & 255));
    else if (k >= 21 && k < 40) v = -2.f * sinf(C * (float)(((k - 20) * w) & 255));
    T5[idx] = (_Float16)v;
  }
}

// ---------- init: weights -> f16, transposed to [mode][o][i] (B^T form) ----------
__global__ __launch_bounds__(256) void init_weights(const float* wr, const float* wi,
                                                    _Float16* Wrt, _Float16* Wit,
                                                    _Float16* WiN) {
  int idx = blockIdx.x * 256 + threadIdx.x;  // 400*64*64 = 1,638,400
  int mode = idx >> 12;
  int o = (idx >> 6) & 63;
  int i = idx & 63;
  int src = (i * 64 + o) * 400 + mode;  // w layout (Cin,Cout,M1,M2)
  float r = wr[src], m = wi[src];
  Wrt[idx] = (_Float16)r;
  Wit[idx] = (_Float16)m;
  WiN[idx] = (_Float16)(-m);
}

// ---------- stage 1: Y = x @ T1  (262144x256x64), write Yt[bi][n][h] ----------
__global__ __launch_bounds__(256) void stage1(const float* x, const _Float16* T1t,
                                              _Float16* Yt) {
  int gw = blockIdx.x * 8 + WAVE;  // 16384 row tiles
  int row0 = gw * 16;
  v8f acc[4] = {};
#pragma unroll
  for (int k0 = 0; k0 < 256; k0 += 32) {
    v16h a = ldA32(x, row0, k0, 256);
#pragma unroll
    for (int nt = 0; nt < 4; ++nt)
      acc[nt] = wmma32(a, ldBT(T1t, nt * 16, k0, 256), acc[nt]);
  }
  int bi = row0 >> 8, h0 = row0 & 255;
  int lane = LANE, n = lane & 15, rb = (lane & 16) ? 8 : 0;
#pragma unroll
  for (int nt = 0; nt < 4; ++nt) {
    v8h pk;
#pragma unroll
    for (int v = 0; v < 8; ++v) pk[v] = (_Float16)acc[nt][v];
    *(v8h*)(Yt + ((bi * 64 + nt * 16 + n) * 256 + h0 + rb)) = pk;  // transposed store
  }
}

// ---------- stage 2: per (b,i): D2 = A2 @ Y_bi  (64x256x64) ----------
__global__ __launch_bounds__(256) void stage2(const _Float16* A2, const _Float16* Yt,
                                              float* D2) {
  int gw = blockIdx.x * 8 + WAVE;  // 16384 = 1024 bi * 4 rt * 4 ct
  int bi = gw >> 4, rt = (gw >> 2) & 3, ct = gw & 3;
  const _Float16* Yb = Yt + bi * 16384;
  v8f acc = {};
#pragma unroll
  for (int k0 = 0; k0 < 256; k0 += 32)
    acc = wmma32(ldA16(A2, rt * 16, k0, 256), ldBT(Yb, ct * 16, k0, 256), acc);
  int lane = LANE, n = lane & 15, rb = (lane & 16) ? 8 : 0;
  float* d = D2 + bi * 4096 + (rt * 16 + rb) * 64 + ct * 16 + n;
#pragma unroll
  for (int v = 0; v < 8; ++v) d[v * 64] = acc[v];
}

// ---------- stage 2b: combine D2 blocks into complex X[mode][b][i] ----------
__global__ __launch_bounds__(256) void combine2(const float* D2, _Float16* Xr,
                                                _Float16* Xi) {
  int idx = blockIdx.x * 256 + threadIdx.x;  // 400*16*64 = 409600
  int mode = idx >> 10;
  int b = (idx >> 6) & 15, i = idx & 63;
  int kx = mode / 20, ky = mode - kx * 20;
  const float* d = D2 + (b * 64 + i) * 4096;
  float xr = d[kx * 64 + ky] + d[(32 + kx) * 64 + 20 + ky];        // C@Yr + S@Yi
  float xi = d[kx * 64 + 20 + ky] - d[(32 + kx) * 64 + ky];        // C@Yi - S@Yr
  Xr[idx] = (_Float16)xr;
  Xi[idx] = (_Float16)xi;
}

// ---------- stage 3: per-mode complex channel mix, G = X * W ----------
__global__ __launch_bounds__(256) void stage3(const _Float16* Xr, const _Float16* Xi,
                                              const _Float16* Wrt, const _Float16* Wit,
                                              const _Float16* WiN, _Float16* Gr,
                                              _Float16* Gi) {
  int gw = blockIdx.x * 8 + WAVE;  // 3200 = 400 modes * 2 parts * 4 coltiles
  int mode = gw >> 3;
  int part = (gw >> 2) & 1, ct = gw & 3;
  const _Float16* A0 = Xr + mode * 1024;
  const _Float16* A1 = Xi + mode * 1024;
  const _Float16* B0 = (part ? Wit : Wrt) + mode * 4096;
  const _Float16* B1 = (part ? Wrt : WiN) + mode * 4096;
  v8f acc = {};
#pragma unroll
  for (int k0 = 0; k0 < 64; k0 += 32)
    acc = wmma32(ldA16(A0, 0, k0, 64), ldBT(B0, ct * 16, k0, 64), acc);
#pragma unroll
  for (int k0 = 0; k0 < 64; k0 += 32)
    acc = wmma32(ldA16(A1, 0, k0, 64), ldBT(B1, ct * 16, k0, 64), acc);
  _Float16* dst = (part ? Gi : Gr) + mode * 1024;
  int lane = LANE, n = lane & 15, rb = (lane & 16) ? 8 : 0;
#pragma unroll
  for (int v = 0; v < 8; ++v) dst[(rb + v) * 64 + ct * 16 + n] = (_Float16)acc[v];
}

// ---------- stage 4: per (b,o): T = A4 @ B4(G), packed inverse-H ----------
static __device__ __forceinline__ _Float16 b4val(const _Float16* Gr, const _Float16* Gi,
                                                 int r, int c, int bo) {
  if (c < 40) {
    int ky = (c < 20) ? c : (c - 20);
    bool imagCol = (c >= 20);
    if (r < 20) {  // cos rows: Gr | Gi
      const _Float16* s = imagCol ? Gi : Gr;
      return s[(r * 20 + ky) * 1024 + bo];
    }
    if (r >= 32 && r < 52) {  // sin rows: -Gi | Gr
      int kx = r - 32;
      if (imagCol) return Gr[(kx * 20 + ky) * 1024 + bo];
      float g = (float)Gi[(kx * 20 + ky) * 1024 + bo];
      return (_Float16)(-g);
    }
  }
  return (_Float16)0.f;
}

__global__ __launch_bounds__(256) void stage4(const _Float16* A4, const _Float16* Gr,
                                              const _Float16* Gi, _Float16* Tb) {
  int gw = blockIdx.x * 8 + WAVE;  // 4096 = 1024 bo * 4 coltiles
  int bo = gw >> 2, ct = gw & 3;
  int lane = LANE, n = lane & 15, rb = (lane & 16) ? 8 : 0;
  int c = ct * 16 + n;
  int koff = (lane & 16) ? 16 : 0;
  v16h bf0, bf1;  // B fragments built in-register, reused over all 16 row tiles
#pragma unroll
  for (int j = 0; j < 16; ++j) bf0[j] = b4val(Gr, Gi, koff + j, c, bo);
#pragma unroll
  for (int j = 0; j < 16; ++j) bf1[j] = b4val(Gr, Gi, 32 + koff + j, c, bo);
  for (int rt = 0; rt < 16; ++rt) {
    v8f acc = {};
    acc = wmma32(ldA16(A4, rt * 16, 0, 64), bf0, acc);
    acc = wmma32(ldA16(A4, rt * 16, 32, 64), bf1, acc);
    _Float16* d = Tb + (bo * 256 + rt * 16 + rb) * 64 + c;
#pragma unroll
    for (int v = 0; v < 8; ++v) d[v * 64] = (_Float16)acc[v];
  }
}

// ---------- stage 5: out = Tb(262144x64) @ T5(64x256), scale 1/(H*W) ----------
__global__ __launch_bounds__(256) void stage5(const _Float16* Tb, const _Float16* T5t,
                                              float* out) {
  int gw = blockIdx.x * 8 + WAVE;  // 65536 = 16384 rowtiles * 4 ngroups
  int rowTile = gw >> 2, ng = gw & 3;
  int row0 = rowTile * 16, n0 = ng * 64;
  v8f acc[4] = {};
#pragma unroll
  for (int k0 = 0; k0 < 64; k0 += 32) {
    v16h a = ldA16(Tb, row0, k0, 64);
#pragma unroll
    for (int nt = 0; nt < 4; ++nt)
      acc[nt] = wmma32(a, ldBT(T5t, n0 + nt * 16, k0, 64), acc[nt]);
  }
  int lane = LANE, n = lane & 15, rb = (lane & 16) ? 8 : 0;
  const float s = 1.0f / 65536.0f;  // 1/(H*W)
#pragma unroll
  for (int nt = 0; nt < 4; ++nt) {
    float* d = out + (size_t)(row0 + rb) * 256 + n0 + nt * 16 + n;
#pragma unroll
    for (int v = 0; v < 8; ++v) d[(size_t)v * 256] = acc[nt][v] * s;
  }
}

extern "C" void kernel_launch(void* const* d_in, const int* in_sizes, int n_in,
                              void* d_out, int out_size, void* d_ws, size_t ws_size,
                              hipStream_t stream) {
  (void)in_sizes; (void)n_in; (void)out_size; (void)ws_size;
  const float* x  = (const float*)d_in[0];
  const float* wr = (const float*)d_in[1];
  const float* wi = (const float*)d_in[2];
  float* out = (float*)d_out;

  // workspace carve (~97 MB total)
  char* p = (char*)d_ws;
  _Float16* Yt  = (_Float16*)p; p += (size_t)1024 * 64 * 256 * 2;   // 33.5 MB
  _Float16* Tb  = (_Float16*)p; p += (size_t)1024 * 256 * 64 * 2;   // 33.5 MB
  float*    D2  = (float*)p;    p += (size_t)1024 * 64 * 64 * 4;    // 16.8 MB
  _Float16* T1t = (_Float16*)p; p += 16384 * 2;
  _Float16* A2t = (_Float16*)p; p += 16384 * 2;
  _Float16* A4t = (_Float16*)p; p += 16384 * 2;
  _Float16* T5t = (_Float16*)p; p += 16384 * 2;
  _Float16* Wrt = (_Float16*)p; p += (size_t)400 * 4096 * 2;        // 3.3 MB
  _Float16* Wit = (_Float16*)p; p += (size_t)400 * 4096 * 2;
  _Float16* WiN = (_Float16*)p; p += (size_t)400 * 4096 * 2;
  _Float16* Xr  = (_Float16*)p; p += (size_t)400 * 1024 * 2;
  _Float16* Xi  = (_Float16*)p; p += (size_t)400 * 1024 * 2;
  _Float16* Gr  = (_Float16*)p; p += (size_t)400 * 1024 * 2;
  _Float16* Gi  = (_Float16*)p; p += (size_t)400 * 1024 * 2;

  init_twiddles<<<64,   256, 0, stream>>>(T1t, A2t, A4t, T5t);
  init_weights <<<6400, 256, 0, stream>>>(wr, wi, Wrt, Wit, WiN);
  stage1       <<<2048, 256, 0, stream>>>(x, T1t, Yt);
  stage2       <<<2048, 256, 0, stream>>>(A2t, Yt, D2);
  combine2     <<<1600, 256, 0, stream>>>(D2, Xr, Xi);
  stage3       <<<400,  256, 0, stream>>>(Xr, Xi, Wrt, Wit, WiN, Gr, Gi);
  stage4       <<<512,  256, 0, stream>>>(A4t, Gr, Gi, Tb);
  stage5       <<<8192, 256, 0, stream>>>(Tb, T5t, out);
}
